// PropPlane_61632780698033
// MI455X (gfx1250) — matrix operand order repaired and checked by reading, not compiled, not run on hardware
//
#include <hip/hip_runtime.h>
#include <hip/hip_bf16.h>

// ---------------------------------------------------------------------------
// PropPlane head for MI455X (gfx1250, wave32).
//   * 3x3 conv (64->24) as implicit GEMM on v_wmma_f32_16x16x32_f16
//   * bilinear 2x feature upsample
//   * fused cosine-affinity / conf-sampling / TGASS-normalize / softmax
// Workspace layout (floats):
//   feaUp : [0,        2457600)                      (B,8,240,320) planar
//   oaA   : [2457600,  4915200)                      (B,8,240,320) aff logits (conv ch16..23)
//   oaI   : [4915200,  9830400)                      (B,H*W,16)    conv ch0..15 interleaved
//   Bpack : halves after that (2*18*32*16 = 18432 _Float16 = 36864 B)
// Total ~39.4 MB of d_ws.
// ---------------------------------------------------------------------------

typedef __attribute__((ext_vector_type(16))) _Float16 v16h;
typedef __attribute__((ext_vector_type(8)))  _Float16 v8h;
typedef __attribute__((ext_vector_type(8)))  float    v8f;

#define B_    4
#define CHG   64
#define H_    240
#define W_    320
#define HW    (H_ * W_)
#define NUM_  8
#define CFEA  8
#define NSTEP 18          // K = 576 = 18 * 32

// -------------------------- B-fragment packing ------------------------------
// K ordering: k = t*64 + c  (t = ky*3+kx tap index, c = input channel).
// B fragment as packed here (contiguous k, not the ISA table order): lane L = column n = L%16;
//   lanes 0-15 hold K = s*32 + 0..15, lanes 16-31 hold K = s*32 + 16..31.
__global__ void pack_b_kernel(const float* __restrict__ w,
                              _Float16* __restrict__ Bpack) {
  int idx = blockIdx.x * blockDim.x + threadIdx.x;
  if (idx >= 2 * NSTEP * 32) return;
  int lane = idx & 31;
  int s    = (idx >> 5) % NSTEP;
  int nt   = idx / (32 * NSTEP);
  int n    = nt * 16 + (lane & 15);
  int kbase = s * 32 + ((lane < 16) ? 0 : 16);
  _Float16* dst = &Bpack[(size_t)idx * 16];
#pragma unroll
  for (int i = 0; i < 16; ++i) {
    int k = kbase + i;
    int t = k >> 6;          // tap
    int c = k & 63;          // input channel
    int ky = t / 3, kx = t % 3;
    float v = (n < 24) ? w[(((size_t)n * CHG + c) * 3 + ky) * 3 + kx] : 0.f;
    dst[i] = (_Float16)v;
  }
}

// -------------------------- 2x bilinear upsample ----------------------------
// jax.image.resize "bilinear": src coord = d*0.5-0.25, edge renormalization
// == index clamping.
__global__ void resize_kernel(const float* __restrict__ src,
                              float* __restrict__ dst) {
  int idx = blockIdx.x * blockDim.x + threadIdx.x;
  if (idx >= B_ * CFEA * HW) return;
  int x  = idx % W_;
  int y  = (idx / W_) % H_;
  int bc = idx / HW;
  const int Hs = H_ / 2, Ws = W_ / 2;
  float sy = y * 0.5f - 0.25f;
  float sx = x * 0.5f - 0.25f;
  float y0f = floorf(sy), x0f = floorf(sx);
  float wy = sy - y0f, wx = sx - x0f;
  int y0 = (int)y0f, x0 = (int)x0f;
  int y0c = min(max(y0, 0), Hs - 1), y1c = min(max(y0 + 1, 0), Hs - 1);
  int x0c = min(max(x0, 0), Ws - 1), x1c = min(max(x0 + 1, 0), Ws - 1);
  const float* p = src + (size_t)bc * Hs * Ws;
  float v = p[y0c * Ws + x0c] * (1.f - wy) * (1.f - wx)
          + p[y0c * Ws + x1c] * (1.f - wy) * wx
          + p[y1c * Ws + x0c] * wy * (1.f - wx)
          + p[y1c * Ws + x1c] * wy * wx;
  dst[idx] = v;
}

// -------------------------- WMMA implicit-GEMM conv -------------------------
// Workgroup: 128 threads = 4 waves; one (b,y) row segment of 64 pixels,
// all 24 (padded 32) output channels. Guidance tile staged in LDS as f16
// with channels contiguous so A-fragment halves are single b128 LDS loads.
// Channels 0..15 (offsets) stored interleaved for the fuse-stage gathers,
// channels 16..23 (aff logits) stored planar.
__global__ __launch_bounds__(128) void conv_wmma_kernel(
    const float* __restrict__ g, const float* __restrict__ bias,
    const _Float16* __restrict__ Bpack,
    float* __restrict__ oaI, float* __restrict__ oaA) {
  __shared__ _Float16 ldsA[3 * 66 * 64];   // 25,344 B

  int bx   = blockIdx.x;
  int tile = bx % 5;                 // W_/64 = 5 tiles per row
  int y    = (bx / 5) % H_;
  int b    = bx / (5 * H_);
  int x0   = tile * 64;
  int tid  = threadIdx.x;

  // Warm L1/L2 with the packed-B fragments (global_prefetch_b8).
  __builtin_prefetch(&Bpack[(size_t)tid * 144], 0, 0);

  // Stage 64ch x 3 rows x 66 cols of guidance (zero-padded) as f16.
  for (int idx = tid; idx < 3 * 66 * 64; idx += 128) {
    int col = idx % 66;
    int tmp = idx / 66;
    int rr  = tmp % 3;
    int c   = tmp / 3;
    int yy = y + rr - 1;
    int xx = x0 + col - 1;
    float v = 0.f;
    if (yy >= 0 && yy < H_ && xx >= 0 && xx < W_)
      v = g[(((size_t)b * CHG + c) * H_ + yy) * W_ + xx];
    ldsA[(rr * 66 + col) * 64 + c] = (_Float16)v;
  }
  __syncthreads();

  int wave = tid >> 5;
  int lane = tid & 31;
  int lrow = lane & 15;
  int hi   = lane >> 4;

  v8f acc0 = {};
  v8f acc1 = {};

#pragma unroll
  for (int t = 0; t < 9; ++t) {
    int dy = t / 3, dx = t % 3;
    const _Float16* arow = &ldsA[(dy * 66 + wave * 16 + lrow + dx) * 64];
#pragma unroll
    for (int half = 0; half < 2; ++half) {
      // A fragment, contiguous k order used here: lanes<16 -> K = cb+{0..15},
      // lanes>=16 offset by 16. cb selects channel half (0 / 32).
      int cb = half * 32 + hi * 16;
      v8h lo = *reinterpret_cast<const v8h*>(arow + cb);
      v8h hv = *reinterpret_cast<const v8h*>(arow + cb + 8);
      v16h a;
#pragma unroll
      for (int e = 0; e < 8; ++e) { a[e] = lo[e]; a[8 + e] = hv[e]; }

      int s = t * 2 + half;
      v16h b0 = *reinterpret_cast<const v16h*>(
          &Bpack[(size_t)(((0 * NSTEP + s) * 32 + lane)) * 16]);
      v16h b1 = *reinterpret_cast<const v16h*>(
          &Bpack[(size_t)(((1 * NSTEP + s) * 32 + lane)) * 16]);

      acc0 = __builtin_amdgcn_wmma_f32_16x16x32_f16(
          false, a, false, b0, (short)0, acc0, false, false);
      acc1 = __builtin_amdgcn_wmma_f32_16x16x32_f16(
          false, a, false, b1, (short)0, acc1, false, false);
    }
  }

  // D layout: lane column n = lrow; VGPR r -> pixel M = hi*8 + r.
  int xbase = x0 + wave * 16 + hi * 8;
  {
    // N-tile 0: channels 0..15 -> interleaved (b, P, 16) for fuse gathers.
    float bb = bias[lrow];
    size_t pbase = (size_t)b * HW + (size_t)y * W_ + xbase;
#pragma unroll
    for (int r = 0; r < 8; ++r)
      oaI[(pbase + r) * 16 + lrow] = acc0[r] + bb;
  }
  if (lrow < 8) {                       // N-tile 1: channels 16..23 planar
    int n = lrow;
    float bb = bias[16 + n];
    float* o = &oaA[(((size_t)b * 8 + n) * H_ + y) * W_ + xbase];
#pragma unroll
    for (int r = 0; r < 8; ++r) o[r] = acc1[r] + bb;
  }
}

// -------------------------- fused affinity stage ----------------------------
__device__ __forceinline__ float tap_zeros(const float* __restrict__ img,
                                           int xi, int yi) {
  if (xi < 0 || xi > W_ - 1 || yi < 0 || yi > H_ - 1) return 0.f;
  return img[yi * W_ + xi];
}

__device__ __forceinline__ float bilin_zeros(const float* __restrict__ img,
                                             float x, float y) {
  float x0f = floorf(x), y0f = floorf(y);
  float wx = x - x0f, wy = y - y0f;
  int x0 = (int)x0f, y0 = (int)y0f;
  if ((unsigned)x0 < (unsigned)(W_ - 1) && (unsigned)y0 < (unsigned)(H_ - 1)) {
    // interior fast path: 4 unpredicated loads
    const float* p0 = img + y0 * W_ + x0;
    float v00 = p0[0], v10 = p0[1];
    float v01 = p0[W_], v11 = p0[W_ + 1];
    float top = fmaf(wx, v10 - v00, v00);
    float bot = fmaf(wx, v11 - v01, v01);
    return fmaf(wy, bot - top, top);
  }
  return tap_zeros(img, x0,     y0)     * (1.f - wy) * (1.f - wx)
       + tap_zeros(img, x0 + 1, y0)     * (1.f - wy) * wx
       + tap_zeros(img, x0,     y0 + 1) * wy * (1.f - wx)
       + tap_zeros(img, x0 + 1, y0 + 1) * wy * wx;
}

__global__ __launch_bounds__(256) void fuse_kernel(
    const float* __restrict__ oaI, const float* __restrict__ oaA,
    const float* __restrict__ feaUp, const float* __restrict__ conf,
    const float* __restrict__ scale_c, float* __restrict__ out) {
  int idx = blockIdx.x * blockDim.x + threadIdx.x;
  if (idx >= B_ * HW) return;
  int P = idx % HW;
  int b = idx / HW;
  int i = P / W_, j = P % W_;

  // Faithful raw-reshape index math: a = channel sampled, P' = 8r+n = location.
  int a = P / 9600;
  int r = P % 9600;

  const float* fe = feaUp + (size_t)b * CFEA * HW;
  const float* fa = fe + (size_t)a * HW;

  float f[8];
#pragma unroll
  for (int n = 0; n < 8; ++n) f[n] = fe[(size_t)n * HW + P];

  // 16 offset channels at this pixel (b128 x4): used for conf sampling and
  // the offset_full output.
  float oP[16];
  {
    const float4* p = reinterpret_cast<const float4*>(
        &oaI[((size_t)b * HW + P) * 16]);
#pragma unroll
    for (int q = 0; q < 4; ++q) {
      float4 v = p[q];
      oP[4 * q + 0] = v.x; oP[4 * q + 1] = v.y;
      oP[4 * q + 2] = v.z; oP[4 * q + 3] = v.w;
    }
  }

  float cw[8] = {0.f, 0.f, 0.f, 0.f, 0.f, 0.f, 0.f, 0.f};
  for (int n = 0; n < 8; ++n) {
    int Pp = r * 8 + n;
    int ip = Pp / W_, jp = Pp % W_;
    float fn = f[n];
    float oS[16];
    {
      const float4* p = reinterpret_cast<const float4*>(
          &oaI[((size_t)b * HW + Pp) * 16]);
#pragma unroll
      for (int q = 0; q < 4; ++q) {
        float4 v = p[q];
        oS[4 * q + 0] = v.x; oS[4 * q + 1] = v.y;
        oS[4 * q + 2] = v.z; oS[4 * q + 3] = v.w;
      }
    }
#pragma unroll
    for (int c = 0; c < 8; ++c) {
      // quirk: both sample coords use the SAME base (ip for c<4 else jp)
      float base = (c < 4) ? (float)ip : (float)jp;
      float px = oS[2 * c] + base;
      float py = oS[2 * c + 1] + base;
      cw[c] += fn * bilin_zeros(fa, px, py);
    }
  }

  float sc = scale_c[0] + 1e-8f;
  const float* cf = conf + (size_t)b * HW;
  float aff[8];
  float asum = 0.f;
#pragma unroll
  for (int c = 0; c < 8; ++c) {
    float raw = oaA[((size_t)b * 8 + c) * HW + P] * cw[c];
    float av = tanhf(raw) / sc;
    // confidence sampling: dy channel 2c + i (row), dx channel 2c+1 + j (col)
    float py2 = oP[2 * c] + (float)i;
    float px2 = oP[2 * c + 1] + (float)j;
    av *= bilin_zeros(cf, px2, py2);
    aff[c] = av;
    asum += fabsf(av);
  }
  asum += 1e-4f;
  asum = fmaxf(asum, 1.0f);
  float ssum = 0.f;
#pragma unroll
  for (int c = 0; c < 8; ++c) { aff[c] /= asum; ssum += aff[c]; }
  float ref = 1.f - ssum;

  float v9[9] = {aff[0], aff[1], aff[2], aff[3], ref,
                 aff[4], aff[5], aff[6], aff[7]};
  float m = v9[0];
#pragma unroll
  for (int c = 1; c < 9; ++c) m = fmaxf(m, v9[c]);
  float es = 0.f;
#pragma unroll
  for (int c = 0; c < 9; ++c) { v9[c] = __expf(v9[c] - m); es += v9[c]; }
  float inv = 1.f / es;

  // outputs: offset_full (B,18,H,W) then aff_full (B,9,H,W)
  float* outOff = out;
  float* outAff = out + (size_t)B_ * 18 * HW;
  size_t ob = (size_t)b * 18 * HW + P;
#pragma unroll
  for (int c = 0; c < 8; ++c) outOff[ob + (size_t)c * HW] = oP[c];
  outOff[ob + (size_t)8 * HW] = 0.f;
  outOff[ob + (size_t)9 * HW] = 0.f;
#pragma unroll
  for (int c = 0; c < 8; ++c) outOff[ob + (size_t)(10 + c) * HW] = oP[8 + c];

  size_t ab = (size_t)b * 9 * HW + P;
#pragma unroll
  for (int c = 0; c < 9; ++c) outAff[ab + (size_t)c * HW] = v9[c] * inv;
}

// ---------------------------------------------------------------------------
extern "C" void kernel_launch(void* const* d_in, const int* in_sizes, int n_in,
                              void* d_out, int out_size, void* d_ws, size_t ws_size,
                              hipStream_t stream) {
  const float* guidance = (const float*)d_in[0];   // (4,64,240,320)
  const float* gtconf   = (const float*)d_in[1];   // (4,1,240,320)
  const float* tgt      = (const float*)d_in[2];   // (4,8,120,160)
  const float* convw    = (const float*)d_in[3];   // (24,64,3,3)
  const float* convb    = (const float*)d_in[4];   // (24,)
  const float* affsc    = (const float*)d_in[5];   // (1,)

  float* ws    = (float*)d_ws;
  float* feaUp = ws;                               // 2,457,600 floats
  float* oaA   = ws + 2457600;                     // 2,457,600 floats
  float* oaI   = ws + 4915200;                     // 4,915,200 floats
  _Float16* Bpack = (_Float16*)(ws + 9830400);
  float* out = (float*)d_out;

  pack_b_kernel<<<(2 * NSTEP * 32 + 127) / 128, 128, 0, stream>>>(convw, Bpack);
  resize_kernel<<<(B_ * CFEA * HW + 255) / 256, 256, 0, stream>>>(tgt, feaUp);
  conv_wmma_kernel<<<5 * H_ * B_, 128, 0, stream>>>(guidance, convb, Bpack,
                                                    oaI, oaA);
  fuse_kernel<<<(B_ * HW + 255) / 256, 256, 0, stream>>>(oaI, oaA, feaUp,
                                                         gtconf, affsc, out);
}
